// GraphAttention_37306085933331
// MI455X (gfx1250) — matrix operand order repaired and checked by reading, not compiled, not run on hardware
//
#include <hip/hip_runtime.h>

#define N 12288
#define DIM 128
#define JT (N / 32)
#define L2E 1.4426950408889634f
#define MASKVAL (-9.0e16f * L2E)

typedef __attribute__((ext_vector_type(16))) _Float16 v16h;
typedef __attribute__((ext_vector_type(8)))  float    v8f;

union U16H { uint4 u[2]; _Float16 h[16]; };
union F16U { float4 v[4]; float f[16]; };
union I16U { int4 v[4]; int i[16]; };

// ---------------------------------------------------------------- W -> f16
__global__ __launch_bounds__(256) void k_prep_w(const float* __restrict__ W,
                                                _Float16* __restrict__ Wh) {
  int i = blockIdx.x * 256 + threadIdx.x;
  if (i < DIM * DIM) Wh[i] = (_Float16)W[i];
}

// --------------------------------------------- e_new = emb @ W^T via WMMA
// One wave per 16x16 output tile. Writes e_new (f32, row-major) and
// VT (f16, transposed: VT[col][row]) for the PV matmul B-fragments.
__global__ __launch_bounds__(256) void k_gemm1(const float* __restrict__ emb,
                                               const _Float16* __restrict__ Wh,
                                               float* __restrict__ enew,
                                               _Float16* __restrict__ VT) {
  const int wv   = threadIdx.x >> 5;
  const int lane = threadIdx.x & 31;
  const int ri   = lane & 15;
  const int hf   = lane >> 4;
  const int ko   = hf * 8;
  const int rowbase = blockIdx.x * 16;
  const int cbase   = wv * 16;

  const float*    arow = emb + (size_t)(rowbase + ri) * DIM;
  const _Float16* brow = Wh  + (size_t)(cbase + ri) * DIM;

  v8f acc = {};
#pragma unroll
  for (int ks = 0; ks < 4; ++ks) {
    const int kb = ks * 32;
    // A fragment: lane holds row (rowbase+ri), K = ko..ko+7 and ko+16..ko+23
    float4 a0 = *(const float4*)(arow + kb + ko);
    float4 a1 = *(const float4*)(arow + kb + ko + 4);
    float4 a2 = *(const float4*)(arow + kb + ko + 16);
    float4 a3 = *(const float4*)(arow + kb + ko + 20);
    v16h A;
    A[0]=(_Float16)a0.x;  A[1]=(_Float16)a0.y;  A[2]=(_Float16)a0.z;  A[3]=(_Float16)a0.w;
    A[4]=(_Float16)a1.x;  A[5]=(_Float16)a1.y;  A[6]=(_Float16)a1.z;  A[7]=(_Float16)a1.w;
    A[8]=(_Float16)a2.x;  A[9]=(_Float16)a2.y;  A[10]=(_Float16)a2.z; A[11]=(_Float16)a2.w;
    A[12]=(_Float16)a3.x; A[13]=(_Float16)a3.y; A[14]=(_Float16)a3.z; A[15]=(_Float16)a3.w;
    // B fragment: B[k][n] = W[cbase+n][kb+k]; lane n=ri, K = ko..ko+7 and ko+16..ko+23
    U16H bu;
    bu.u[0] = *(const uint4*)(brow + kb + ko);
    bu.u[1] = *(const uint4*)(brow + kb + ko + 16);
    v16h B;
#pragma unroll
    for (int t = 0; t < 16; ++t) B[t] = bu.h[t];
    acc = __builtin_amdgcn_wmma_f32_16x16x32_f16(false, A, false, B, (short)0,
                                                 acc, false, false);
  }
  // C layout: vreg r -> row (rowbase + r + 8*hf), col = cbase + ri
  U16H vt;
#pragma unroll
  for (int r = 0; r < 8; ++r) {
    enew[(size_t)(rowbase + r + 8 * hf) * DIM + cbase + ri] = acc[r];
    vt.h[r] = (_Float16)acc[r];
  }
  *(uint4*)(VT + (size_t)(cbase + ri) * N + rowbase + 8 * hf) = vt.u[0];
}

// ------------------------------------------ src/tgt scores (one wave/row)
__global__ __launch_bounds__(256) void k_scores(const float* __restrict__ enew,
                                                const float* __restrict__ a,
                                                float* __restrict__ src,
                                                float* __restrict__ tgt) {
  int row  = (blockIdx.x * 256 + threadIdx.x) >> 5;
  int lane = threadIdx.x & 31;
  if (row >= N) return;
  const float* r = enew + (size_t)row * DIM;
  float4 v  = *(const float4*)(r + lane * 4);
  float4 w1 = *(const float4*)(a + lane * 4);
  float4 w2 = *(const float4*)(a + DIM + lane * 4);
  float s1 = v.x * w1.x + v.y * w1.y + v.z * w1.z + v.w * w1.w;
  float s2 = v.x * w2.x + v.y * w2.y + v.z * w2.z + v.w * w2.w;
#pragma unroll
  for (int off = 16; off > 0; off >>= 1) {
    s1 += __shfl_xor(s1, off, 32);
    s2 += __shfl_xor(s2, off, 32);
  }
  if (lane == 0) { src[row] = s1; tgt[row] = s2; }
}

// --------------- fused masked-softmax attention * V (flash-style, WMMA PV)
__global__ __launch_bounds__(256) void k_attn(const int* __restrict__ adj,
                                              const float* __restrict__ srcv,
                                              const float* __restrict__ tgtv,
                                              const _Float16* __restrict__ VT,
                                              float* __restrict__ out) {
  __shared__ float lds_acc[16 * DIM];
  __shared__ float lds_m[8][16];
  __shared__ float lds_l[8][16];

  const int wv   = threadIdx.x >> 5;
  const int lane = threadIdx.x & 31;
  const int ri   = lane & 15;     // this lane's row within the 16-row block
  const int hf   = lane >> 4;     // half-wave select
  const int ko   = hf * 8;        // A-fragment K offset for this half
  const int rowbase = blockIdx.x * 16;

  const float srow   = srcv[rowbase + ri];
  const int* adjrow  = adj + (size_t)(rowbase + ri) * N + ko;
  const float* tgtk  = tgtv + ko;

  float m = -__builtin_inff();
  float l = 0.f;
  v8f acc[8] = {};

  for (int jt = wv; jt < JT; jt += 8) {
    const int jb = jt * 32;
    if (jt + 8 < JT) __builtin_prefetch((const void*)(adjrow + jb + 256), 0, 1);

    I16U AD;
    AD.v[0] = *(const int4*)(adjrow + jb);
    AD.v[1] = *(const int4*)(adjrow + jb + 4);
    AD.v[2] = *(const int4*)(adjrow + jb + 16);
    AD.v[3] = *(const int4*)(adjrow + jb + 20);
    F16U TG;
    TG.v[0] = *(const float4*)(tgtk + jb);
    TG.v[1] = *(const float4*)(tgtk + jb + 4);
    TG.v[2] = *(const float4*)(tgtk + jb + 16);
    TG.v[3] = *(const float4*)(tgtk + jb + 20);

    // scores in exp2 domain: t = leaky_relu(src+tgt)*log2(e), masked
    float t[16];
#pragma unroll
    for (int k = 0; k < 16; ++k) {
      float x = srow + TG.f[k];
      float e = (x > 0.f) ? x : 0.01f * x;
      t[k] = (AD.i[k] > 0) ? e * L2E : MASKVAL;
    }
    float tmax = t[0];
#pragma unroll
    for (int k = 1; k < 16; ++k) tmax = fmaxf(tmax, t[k]);
    tmax = fmaxf(tmax, __shfl_xor(tmax, 16, 32));   // combine the two halves (same row)

    const float mold  = m;
    const float mnew  = fmaxf(m, tmax);
    const float scale = __builtin_amdgcn_exp2f(mold - mnew);

    v16h A;
    float psum = 0.f;
#pragma unroll
    for (int k = 0; k < 16; ++k) {
      float p = __builtin_amdgcn_exp2f(t[k] - mnew);
      psum += p;
      A[k] = (_Float16)p;
    }
    psum += __shfl_xor(psum, 16, 32);
    l = l * scale + psum;
    m = mnew;

    // rescale accumulators only if any row's max moved (wave-uniform branch)
    if (__builtin_amdgcn_ballot_w32(mnew > mold) != 0u) {
#pragma unroll
      for (int r = 0; r < 8; ++r) {
        float f = __shfl(scale, hf ? (r + 8) : r, 32);  // scale for row r+8*hf
#pragma unroll
        for (int s = 0; s < 8; ++s) acc[s][r] *= f;
      }
    }

    // PV: 8 col-slabs, B[k][n] = VT[16s+n][jb+k]; lane n=ri, K=ko..ko+7, ko+16..ko+23
    const _Float16* vb = VT + jb + ko;
#pragma unroll
    for (int s = 0; s < 8; ++s) {
      const _Float16* p = vb + (size_t)(s * 16 + ri) * N;
      U16H bu;
      bu.u[0] = *(const uint4*)(p);
      bu.u[1] = *(const uint4*)(p + 16);
      v16h B;
#pragma unroll
      for (int tt = 0; tt < 16; ++tt) B[tt] = bu.h[tt];
      acc[s] = __builtin_amdgcn_wmma_f32_16x16x32_f16(false, A, false, B, (short)0,
                                                      acc[s], false, false);
    }
  }

  // ---- merge the 8 waves' partial softmax states ----
  if (hf == 0) lds_m[wv][ri] = m;
  for (int e = threadIdx.x; e < 16 * DIM; e += 256) lds_acc[e] = 0.f;
  __syncthreads();

  float M = -__builtin_inff();
#pragma unroll
  for (int w = 0; w < 8; ++w) M = fmaxf(M, lds_m[w][ri]);
  const float fsc = __builtin_amdgcn_exp2f(m - M);
  if (hf == 0) lds_l[wv][ri] = l * fsc;

#pragma unroll
  for (int r = 0; r < 8; ++r) {
    float f = __shfl(fsc, hf ? (r + 8) : r, 32);
#pragma unroll
    for (int s = 0; s < 8; ++s)
      atomicAdd(&lds_acc[(r + 8 * hf) * DIM + s * 16 + ri], acc[s][r] * f);
  }
  __syncthreads();

  for (int e = threadIdx.x; e < 16 * DIM; e += 256) {
    int row = e >> 7;
    float L = 0.f;
#pragma unroll
    for (int w = 0; w < 8; ++w) L += lds_l[w][row];
    float o = lds_acc[e] / L;
    out[(size_t)rowbase * DIM + e] = fmaxf(o, 0.f);
  }
}

// ------------------------------------------------------------------ launch
extern "C" void kernel_launch(void* const* d_in, const int* in_sizes, int n_in,
                              void* d_out, int out_size, void* d_ws, size_t ws_size,
                              hipStream_t stream) {
  (void)in_sizes; (void)n_in; (void)out_size; (void)ws_size;
  const float* emb = (const float*)d_in[0];
  const int*   adj = (const int*)d_in[1];
  const float* W   = (const float*)d_in[2];
  const float* a   = (const float*)d_in[3];
  float* out = (float*)d_out;

  char* ws = (char*)d_ws;
  _Float16* Wh = (_Float16*)ws;                                   // 32 KB
  _Float16* VT = (_Float16*)(ws + 32768);                         // 3 MB
  float* enew  = (float*)(ws + 32768 + (size_t)DIM * N * 2);      // 6 MB
  float* src   = (float*)(ws + 32768 + (size_t)DIM * N * 2 + (size_t)N * DIM * 4);
  float* tgt   = src + N;

  k_prep_w<<<(DIM * DIM + 255) / 256, 256, 0, stream>>>(W, Wh);
  k_gemm1 <<<N / 16, 256, 0, stream>>>(emb, Wh, enew, VT);
  k_scores<<<N / 8, 256, 0, stream>>>(enew, a, src, tgt);
  k_attn  <<<N / 16, 256, 0, stream>>>(adj, src, tgt, VT, out);
}
